// LearnableVectorQuantization_51634096832640
// MI455X (gfx1250) — matrix operand order repaired and checked by reading, not compiled, not run on hardware
//
#include <hip/hip_runtime.h>

// ---------------------------------------------------------------------------
// Vector quantization argmin for MI455X (gfx1250, wave32, WMMA + TDM).
//   indices[n] = argmin_k || x[n] - v[k] || == argmin_k ( v2[k] - 2<x[n],v[k]> )
// GEMM via v_wmma_f32_16x16x32_bf16 with hi/lo bf16 split (3 passes ~ fp32).
// Codebook tiles are streamed LDS-side by the Tensor Data Mover
// (tensor_load_to_lds + s_wait_tensorcnt), double-buffered so DMA overlaps
// the WMMA stream. Falls back to plain loads if the builtin is absent.
// ---------------------------------------------------------------------------

typedef __attribute__((ext_vector_type(16))) __bf16        v16bf;
typedef __attribute__((ext_vector_type(8)))  float         v8f;
typedef __attribute__((ext_vector_type(4)))  float         f32x4;
typedef __attribute__((ext_vector_type(4)))  unsigned int  u32x4;
typedef __attribute__((ext_vector_type(2)))  unsigned int  u32x2;
typedef __attribute__((ext_vector_type(8)))  int           i32x8;
typedef __attribute__((ext_vector_type(4)))  int           i32x4;

constexpr int QD   = 512;    // feature dim
constexpr int NB   = 4096;   // codebook entries
constexpr int NPTS = 32768;  // query rows

constexpr int ROWS = 128;    // x rows per workgroup (8 waves x 16 rows)
constexpr int AP   = 520;    // padded LDS stride for A rows (bf16 elems)
constexpr int BP   = 72;     // padded LDS stride for B rows (64 data + 8 pad)
constexpr int CCH  = 64;     // codewords per N-chunk (4 WMMA col tiles)
constexpr int DT   = 64;     // K-extent of a staged B tile (2 WMMA K-steps)
constexpr int TILES = (NB / CCH) * (QD / DT);          // 512
constexpr int BUFB  = CCH * BP * 2;                    // bytes per B buffer (hi or lo)

#if defined(__has_builtin)
#if __has_builtin(__builtin_amdgcn_tensor_load_to_lds) && \
    __has_builtin(__builtin_amdgcn_s_wait_tensorcnt)
#define USE_TDM 1
#endif
#endif
#ifndef USE_TDM
#define USE_TDM 0
#endif
#if __has_include(<hip/amd_detail/amd_gfx1250_TDM.h>)
#define TDM_6ARG 1
#else
#define TDM_6ARG 0
#endif

union FragBF {                // POD union: raw dwords <-> WMMA bf16 fragment
  u32x4 u[2];
  v16bf v;
};

// ---------------------------------------------------------------------------
// Prep: codebook fp32 -> bf16 hi/lo split + per-row squared norms.
// ---------------------------------------------------------------------------
__global__ __launch_bounds__(256) void vq_prep(const float* __restrict__ vecs,
                                               __bf16* __restrict__ vhi,
                                               __bf16* __restrict__ vlo,
                                               float* __restrict__ v2) {
  const int wave = threadIdx.x >> 5;
  const int lane = threadIdx.x & 31;
  const int row  = blockIdx.x * 8 + wave;

  const float* src = vecs + (size_t)row * QD + lane * 16;
  union { __bf16 b[16]; u32x4 u[2]; } H, L;
  float sum = 0.0f;
#pragma unroll
  for (int i = 0; i < 16; ++i) {
    float xv = src[i];
    sum = fmaf(xv, xv, sum);
    __bf16 h = (__bf16)xv;
    H.b[i] = h;
    L.b[i] = (__bf16)(xv - (float)h);
  }
  u32x4* dh = (u32x4*)(vhi + (size_t)row * QD + lane * 16);
  dh[0] = H.u[0];
  dh[1] = H.u[1];
  u32x4* dl = (u32x4*)(vlo + (size_t)row * QD + lane * 16);
  dl[0] = L.u[0];
  dl[1] = L.u[1];

#pragma unroll
  for (int off = 16; off > 0; off >>= 1) sum += __shfl_xor(sum, off, 32);
  if (lane == 0) v2[row] = sum;
}

#if USE_TDM
// Issue one TDM 2D tile load: 64 codewords x 128B (64 bf16) into LDS with a
// 16B pad after every 128B row (padded LDS row stride = 144B = 72 bf16).
__device__ __forceinline__ void tdm_issue(const __bf16* gsrc, unsigned ldsOff) {
  unsigned long long ga = (unsigned long long)gsrc;
  u32x4 g0;
  g0[0] = 1u;                                   // count=1, user descriptor
  g0[1] = ldsOff;                               // LDS byte address of tile
  g0[2] = (unsigned)ga;                         // global_addr[31:0]
  g0[3] = (unsigned)((ga >> 32) & 0x01FFFFFFu)  // global_addr[56:32]
          | (2u << 30);                         // type = 2 ("image")
  i32x8 g1;
  g1[0] = (int)((3u << 16)      // data_size   = 8B units
                | (1u << 20)    // pad_enable
                | (4u << 22)    // pad_interval: 2^(4+1)=32 dwords = 128B
                | (3u << 25));  // pad_amount : 4 dwords = 16B
  g1[1] = (int)(128u << 16);    // tensor_dim0 = 128 units (512 bf16 row)
  g1[2] = (int)(4096u << 16);   // tensor_dim1 = 4096 codewords
  g1[3] = (int)(16u << 16);     // tile_dim0   = 16 units (128B = 64 bf16)
  g1[4] = 64;                   // tile_dim1   = 64 codewords (tile_dim2 = 0)
  g1[5] = 128;                  // tensor_dim0_stride = 128 units (1024B)
  g1[6] = 0;                    // dim1_stride unused for 2D tile
  g1[7] = 0;
  i32x4 gz = (i32x4){0, 0, 0, 0};
#if TDM_6ARG
  __builtin_amdgcn_tensor_load_to_lds(g0, g1, gz, gz,
                                      (i32x8){0, 0, 0, 0, 0, 0, 0, 0}, 0);
#else
  __builtin_amdgcn_tensor_load_to_lds(g0, g1, gz, gz, 0);
#endif
}
#endif

// ---------------------------------------------------------------------------
// Main kernel.
// ---------------------------------------------------------------------------
__global__ __launch_bounds__(256) void vq_main(const float* __restrict__ x,
                                               const __bf16* __restrict__ vhi,
                                               const __bf16* __restrict__ vlo,
                                               const float* __restrict__ v2,
                                               float* __restrict__ out) {
  __shared__ __bf16 sAh[ROWS * AP];        // 133,120 B
  __shared__ __bf16 sAl[ROWS * AP];        // 133,120 B
  __shared__ __bf16 sBh[2][CCH * BP];      //  18,432 B (double buffer)
  __shared__ __bf16 sBl[2][CCH * BP];      //  18,432 B
  __shared__ float  redV[8 * 16 * 16];     //   8,192 B
  __shared__ int    redI[8 * 16 * 16];     //   8,192 B  total 319,488 B

  const int tid     = threadIdx.x;
  const int wave    = tid >> 5;
  const int lane    = tid & 31;
  const int laneCol = lane & 15;
  const int hiHalf  = lane >> 4;
  const int rowBase = blockIdx.x * ROWS;

  // ---- Phase 1: stage this WG's 128x512 fp32 x block as bf16 hi/lo ----
  {
    const f32x4* xg = (const f32x4*)(x + (size_t)rowBase * QD);
#pragma unroll 4
    for (int i = 0; i < (ROWS * QD / 4) / 256; ++i) {
      const int lin = tid + i * 256;
      const int row = lin >> 7;
      const int col = (lin & 127) << 2;
      f32x4 f = xg[lin];
      union { __bf16 b[4]; u32x2 u; } Hh, Ll;
#pragma unroll
      for (int j = 0; j < 4; ++j) {
        float  e = f[j];
        __bf16 h = (__bf16)e;
        Hh.b[j]  = h;
        Ll.b[j]  = (__bf16)(e - (float)h);
      }
      *(u32x2*)&sAh[row * AP + col] = Hh.u;
      *(u32x2*)&sAl[row * AP + col] = Ll.u;
    }
  }
  __syncthreads();

  float best[8];
  int   bidx[8];
#pragma unroll
  for (int r = 0; r < 8; ++r) { best[r] = 3.4e38f; bidx[r] = 0; }

  // A fragment (16-bit A 16x32 ISA layout): elem i -> K = i + (i>=8?8:0) + 8*hiHalf
  const int aOff = (wave * 16 + laneCol) * AP + (hiHalf ? 8 : 0);
  // B fragment (32x16): elem i -> K = i + (i>=8?8:0) + 8*hiHalf, col = lane&15
  const int bOff = laneCol * BP + (hiHalf ? 8 : 0);

  v8f acc[4];
#pragma unroll
  for (int t4 = 0; t4 < 4; ++t4)
    acc[t4] = (v8f){0.f, 0.f, 0.f, 0.f, 0.f, 0.f, 0.f, 0.f};

#if USE_TDM
  const unsigned sBhOff = (unsigned)(unsigned long long)&sBh[0][0];
  const unsigned sBlOff = (unsigned)(unsigned long long)&sBl[0][0];
  if (wave == 0) {   // prologue: DMA tile 0 into buffer 0
    tdm_issue(vhi, sBhOff);
    tdm_issue(vlo, sBlOff);
  }
#endif
  int buf = 0;

  for (int t = 0; t < TILES; ++t) {
    const int nc   = t >> 3;          // chunk of 64 codewords
    const int dblk = t & 7;           // 64-wide K block within the chunk
    (void)dblk;

#if USE_TDM
    if (wave == 0) {
      if (t + 1 < TILES) {
        const int nn = (t + 1) >> 3, nd = (t + 1) & 7;
        const size_t goff = (size_t)(nn * CCH) * QD + nd * DT;
        const unsigned lo  = (unsigned)((buf ^ 1) * BUFB);
        tdm_issue(vhi + goff, sBhOff + lo);
        tdm_issue(vlo + goff, sBlOff + lo);
        __builtin_amdgcn_s_wait_tensorcnt(2);   // tile t landed
      } else {
        __builtin_amdgcn_s_wait_tensorcnt(0);
      }
    }
    __syncthreads();                  // tile t visible to all waves
#else
    __syncthreads();                  // previous tile fully consumed
    {                                 // synchronous staging fallback
      const int    c    = tid >> 2;          // 0..63
      const int    doff = (tid & 3) * 16;    // 0,16,32,48 (bf16 elems)
      const size_t g = (size_t)(nc * CCH + c) * QD + dblk * DT + doff;
      *(u32x4*)&sBh[0][c * BP + doff]     = *(const u32x4*)(vhi + g);
      *(u32x4*)&sBh[0][c * BP + doff + 8] = *(const u32x4*)(vhi + g + 8);
      *(u32x4*)&sBl[0][c * BP + doff]     = *(const u32x4*)(vlo + g);
      *(u32x4*)&sBl[0][c * BP + doff + 8] = *(const u32x4*)(vlo + g + 8);
    }
    __syncthreads();
#endif

    const __bf16* bh0 = &sBh[buf][0];
    const __bf16* bl0 = &sBl[buf][0];
    const int     dBase = (t & 7) * DT;

#pragma unroll
    for (int sub = 0; sub < 2; ++sub) {           // two 32-wide K steps
      const int ab = aOff + dBase + sub * 32;
      FragBF ah, al;
      ah.u[0] = *(const u32x4*)&sAh[ab];
      ah.u[1] = *(const u32x4*)&sAh[ab + 16];
      al.u[0] = *(const u32x4*)&sAl[ab];
      al.u[1] = *(const u32x4*)&sAl[ab + 16];

#pragma unroll
      for (int t4 = 0; t4 < 4; ++t4) {
        const int bb = bOff + t4 * 16 * BP + sub * 32;
        FragBF bh, bl;
        bh.u[0] = *(const u32x4*)(bh0 + bb);
        bh.u[1] = *(const u32x4*)(bh0 + bb + 16);
        bl.u[0] = *(const u32x4*)(bl0 + bb);
        bl.u[1] = *(const u32x4*)(bl0 + bb + 16);
        acc[t4] = __builtin_amdgcn_wmma_f32_16x16x32_bf16(
            false, ah.v, false, bh.v, (short)0, acc[t4], false, false);
        acc[t4] = __builtin_amdgcn_wmma_f32_16x16x32_bf16(
            false, ah.v, false, bl.v, (short)0, acc[t4], false, false);
        acc[t4] = __builtin_amdgcn_wmma_f32_16x16x32_bf16(
            false, al.v, false, bh.v, (short)0, acc[t4], false, false);
      }
    }

    if ((t & 7) == 7) {               // chunk complete: fold running argmin
      const int cBase = nc * CCH;
#pragma unroll
      for (int t4 = 0; t4 < 4; ++t4) {
        const int   col = cBase + t4 * 16 + laneCol;
        const float v2c = v2[col];
#pragma unroll
        for (int r = 0; r < 8; ++r) {
          float val = fmaf(-2.0f, acc[t4][r], v2c);   // v2 - 2*dot
          if (val < best[r]) { best[r] = val; bidx[r] = col; }
        }
        acc[t4] = (v8f){0.f, 0.f, 0.f, 0.f, 0.f, 0.f, 0.f, 0.f};
      }
    }

#if USE_TDM
    __syncthreads();                  // all waves done with buf before reuse
    buf ^= 1;
#endif
  }

  // ---- Cross-lane argmin: C/D row M = r + 8*hiHalf; 16 lanes share a row ----
#pragma unroll
  for (int r = 0; r < 8; ++r) {
    const int o = (wave * 16 + r + hiHalf * 8) * 16 + laneCol;
    redV[o] = best[r];
    redI[o] = bidx[r];
  }
  __syncthreads();

  if (tid < ROWS) {
    const int o  = tid * 16;
    float     bv = redV[o];
    int       bi = redI[o];
#pragma unroll
    for (int j = 1; j < 16; ++j) {
      float v  = redV[o + j];
      int   i2 = redI[o + j];
      if (v < bv || (v == bv && i2 < bi)) { bv = v; bi = i2; }
    }
    out[rowBase + tid] = (float)bi;
  }
}

// ---------------------------------------------------------------------------
// Launch. Workspace: vhi (4 MB) | vlo (4 MB) | v2 (16 KB).
// ---------------------------------------------------------------------------
extern "C" void kernel_launch(void* const* d_in, const int* in_sizes, int n_in,
                              void* d_out, int out_size, void* d_ws,
                              size_t ws_size, hipStream_t stream) {
  const float* x    = (const float*)d_in[0];   // [32768, 512] f32
  const float* vecs = (const float*)d_in[1];   // [4096, 512]  f32
  float*       out  = (float*)d_out;           // [32768] indices as f32

  __bf16* vhi = (__bf16*)d_ws;
  __bf16* vlo = vhi + (size_t)NB * QD;
  float*  v2  = (float*)(vlo + (size_t)NB * QD);

  vq_prep<<<NB / 8, 256, 0, stream>>>(vecs, vhi, vlo, v2);
  vq_main<<<NPTS / ROWS, 256, 0, stream>>>(x, vhi, vlo, v2, out);
}
